// RobustDriftGNN_78374563217611
// MI455X (gfx1250) — matrix operand order repaired and checked by reading, not compiled, not run on hardware
//
#include <hip/hip_runtime.h>
#include <hip/hip_bf16.h>
#include <math.h>

// ---------------- problem constants (match reference) ----------------
#define N_NODES 50000
#define N_EDGES 800000
#define E2      (N_EDGES + N_NODES)   // edges + self loops = 850000
#define G_GR    50
#define F_IN    64
#define HID     128
#define EMB     64
#define H1      8
#define NCLS    6
#define NEG_SLOPE 0.2f
#define BN_EPS    1e-5f

typedef __attribute__((ext_vector_type(16))) _Float16 v16h;
typedef __attribute__((ext_vector_type(8)))  _Float16 v8h;
typedef __attribute__((ext_vector_type(8)))  float    v8f;

// ---------------- small device helpers ----------------
__device__ __forceinline__ float lrelu(float g) { return g > 0.f ? g : NEG_SLOPE * g; }

__device__ __forceinline__ float atomicMaxF(float* addr, float value) {
    float old = *addr;
    while (old < value) {
        unsigned int assumed = __float_as_uint(old);
        unsigned int r = atomicCAS((unsigned int*)addr, assumed, __float_as_uint(value));
        old = __uint_as_float(r);
        if (r == assumed) break;
    }
    return old;
}

__device__ __forceinline__ void edge_ends(int e, const int* __restrict__ src,
                                          const int* __restrict__ dst, int& s, int& d) {
    if (e < N_EDGES) { s = src[e]; d = dst[e]; }
    else             { s = d = e - N_EDGES; }   // self loop
}

// ---------------- generic fill (graph-capture-safe init) ----------------
__global__ void k_fill(float* __restrict__ p, float v, size_t n) {
    size_t i = (size_t)blockIdx.x * blockDim.x + threadIdx.x;
    if (i < n) p[i] = v;
}

// ---------------- self-loop attr (fill_value='mean') ----------------
__global__ void k_deg_sum(const int* __restrict__ dst, const float* __restrict__ eattr,
                          float* __restrict__ deg, float* __restrict__ easum,
                          float* __restrict__ ea2) {
    int e = blockIdx.x * blockDim.x + threadIdx.x;
    if (e >= N_EDGES) return;
    int d = dst[e];
    float a = eattr[e];
    atomicAdd(&deg[d], 1.0f);
    atomicAdd(&easum[d], a);
    ea2[e] = a;
}

__global__ void k_loop_attr(const float* __restrict__ deg, const float* __restrict__ easum,
                            float* __restrict__ ea2) {
    int i = blockIdx.x * blockDim.x + threadIdx.x;
    if (i >= N_NODES) return;
    float dg = deg[i];
    ea2[N_EDGES + i] = dg > 0.f ? easum[i] / fmaxf(dg, 1.f) : 0.f;
}

// ---------------- WMMA dense projection: Y = X@W + b  (f16 in, f32 acc) ----
// Block covers 64 rows (4 row-tiles of 16) x M cols; one wave per 16-col tile.
// W is stored TRANSPOSED in LDS so each lane's B fragment is two runs of 8 contiguous
// halves -> two ds_load_b128.  Rows padded to K+8 halves: row stride is
// 16B-aligned (144B / 272B) and spreads the 16 lanes across distinct banks.
template<int K, int M>
__global__ __launch_bounds__(M * 2)
void k_gemm_wmma(const float* __restrict__ X, const float* __restrict__ W,
                 const float* __restrict__ bias, float* __restrict__ Y, int nrows) {
    constexpr int KP = K + 8;
    __shared__ _Float16 sX[64][KP];    // X tile, row-major
    __shared__ _Float16 sWt[M][KP];    // W transposed: sWt[n][k]
    const int tid = threadIdx.x, nth = blockDim.x;
    const int row0 = blockIdx.x * 64;

    for (int i = tid; i < 64 * K; i += nth) {
        int r = i / K, k = i % K;
        int row = row0 + r;
        sX[r][k] = (row < nrows) ? (_Float16)X[(size_t)row * K + k] : (_Float16)0.0f;
    }
    for (int i = tid; i < K * M; i += nth) {
        int k = i / M, n = i % M;
        sWt[n][k] = (_Float16)W[i];
    }
    __syncthreads();

    const int wave = tid >> 5;       // which 16-column tile
    const int lane = tid & 31;
    const int grp  = lane >> 4;      // lane group (0/1)
    const int l16  = lane & 15;      // A: M row;  B/C/D: N column
    const int col  = wave * 16 + l16;

    v8f zero = {0.f, 0.f, 0.f, 0.f, 0.f, 0.f, 0.f, 0.f};
    v8f acc[4] = {zero, zero, zero, zero};

    #pragma unroll
    for (int kk = 0; kk < K; kk += 32) {
        // B fragment: 32x16 f16.  elem h(0..7)->K = kk+8*grp+h ; h(8..15)->K = kk+16+8*grp+(h-8), N = col
        const _Float16* pb = &sWt[col][kk + 8 * grp];
        v8h blo = *(const v8h*)pb;
        v8h bhi = *(const v8h*)(pb + 16);
        v16h b = __builtin_shufflevector(blo, bhi, 0, 1, 2, 3, 4, 5, 6, 7,
                                         8, 9, 10, 11, 12, 13, 14, 15);
        #pragma unroll
        for (int t = 0; t < 4; ++t) {
            // A fragment: elem h(0..7)->K = kk+8*grp+h ; h(8..15)->K = kk+16+8*grp+(h-8)
            const _Float16* pa = &sX[t * 16 + l16][kk + 8 * grp];
            v8h alo = *(const v8h*)pa;
            v8h ahi = *(const v8h*)(pa + 16);
            v16h a = __builtin_shufflevector(alo, ahi, 0, 1, 2, 3, 4, 5, 6, 7,
                                             8, 9, 10, 11, 12, 13, 14, 15);
            acc[t] = __builtin_amdgcn_wmma_f32_16x16x32_f16(false, a, false, b,
                                                            (short)0, acc[t], false, false);
        }
    }
    // C/D layout: VGPR r -> M = r + 8*grp, N = col
    float bv = bias[col];
    #pragma unroll
    for (int t = 0; t < 4; ++t) {
        #pragma unroll
        for (int r = 0; r < 8; ++r) {
            int row = row0 + t * 16 + r + 8 * grp;
            if (row < nrows) Y[(size_t)row * M + col] = acc[t][r] + bv;
        }
    }
}

// ---------------- GATv2 layer 1 (8 heads x 16 ch) ----------------
__global__ void k_logits1(const float* __restrict__ xl, const float* __restrict__ xr,
                          const float* __restrict__ ea2, const int* __restrict__ src,
                          const int* __restrict__ dst, const float* __restrict__ We,
                          const float* __restrict__ att, float* __restrict__ alpha,
                          float* __restrict__ m) {
    int idx = blockIdx.x * blockDim.x + threadIdx.x;   // e*8 + h
    if (idx >= E2 * H1) return;
    int e = idx >> 3, h = idx & 7;
    int s, d; edge_ends(e, src, dst, s, d);
    float a = ea2[e];
    const float* pl = xl + (size_t)s * HID + h * 16;
    const float* pr = xr + (size_t)d * HID + h * 16;
    const float* pw = We + h * 16;
    const float* pa = att + h * 16;
    float acc = 0.f;
    #pragma unroll
    for (int c = 0; c < 16; ++c)
        acc += pa[c] * lrelu(pl[c] + pr[c] + a * pw[c]);
    alpha[idx] = acc;
    atomicMaxF(&m[(size_t)d * H1 + h], acc);
}

__global__ void k_exp1(const int* __restrict__ src, const int* __restrict__ dst,
                       const float* __restrict__ m, float* __restrict__ alpha,
                       float* __restrict__ ssum) {
    int idx = blockIdx.x * blockDim.x + threadIdx.x;
    if (idx >= E2 * H1) return;
    int e = idx >> 3, h = idx & 7;
    int s, d; edge_ends(e, src, dst, s, d);
    float v = expf(alpha[idx] - m[(size_t)d * H1 + h]);
    alpha[idx] = v;
    atomicAdd(&ssum[(size_t)d * H1 + h], v);
}

__global__ void k_scatter1(const int* __restrict__ src, const int* __restrict__ dst,
                           const float* __restrict__ alpha, const float* __restrict__ ssum,
                           const float* __restrict__ xl, float* __restrict__ out) {
    size_t idx = (size_t)blockIdx.x * blockDim.x + threadIdx.x;  // e*128 + c
    if (idx >= (size_t)E2 * HID) return;
    int e = (int)(idx >> 7), c = (int)(idx & 127), h = c >> 4;
    int s, d; edge_ends(e, src, dst, s, d);
    float coeff = alpha[(size_t)e * H1 + h] / ssum[(size_t)d * H1 + h];
    atomicAdd(&out[(size_t)d * HID + c], xl[(size_t)s * HID + c] * coeff);
}

// ---------------- GATv2 layer 2 (1 head x 64 ch) ----------------
__global__ void k_logits2(const float* __restrict__ xl, const float* __restrict__ xr,
                          const float* __restrict__ ea2, const int* __restrict__ src,
                          const int* __restrict__ dst, const float* __restrict__ We,
                          const float* __restrict__ att, float* __restrict__ alpha,
                          float* __restrict__ m) {
    int e = blockIdx.x * blockDim.x + threadIdx.x;
    if (e >= E2) return;
    int s, d; edge_ends(e, src, dst, s, d);
    float a = ea2[e];
    const float* pl = xl + (size_t)s * EMB;
    const float* pr = xr + (size_t)d * EMB;
    float acc = 0.f;
    #pragma unroll 8
    for (int c = 0; c < EMB; ++c)
        acc += att[c] * lrelu(pl[c] + pr[c] + a * We[c]);
    alpha[e] = acc;
    atomicMaxF(&m[d], acc);
}

__global__ void k_exp2(const int* __restrict__ src, const int* __restrict__ dst,
                       const float* __restrict__ m, float* __restrict__ alpha,
                       float* __restrict__ ssum) {
    int e = blockIdx.x * blockDim.x + threadIdx.x;
    if (e >= E2) return;
    int s, d; edge_ends(e, src, dst, s, d);
    float v = expf(alpha[e] - m[d]);
    alpha[e] = v;
    atomicAdd(&ssum[d], v);
}

__global__ void k_scatter2(const int* __restrict__ src, const int* __restrict__ dst,
                           const float* __restrict__ alpha, const float* __restrict__ ssum,
                           const float* __restrict__ xl, float* __restrict__ out) {
    size_t idx = (size_t)blockIdx.x * blockDim.x + threadIdx.x;  // e*64 + c
    if (idx >= (size_t)E2 * EMB) return;
    int e = (int)(idx >> 6), c = (int)(idx & 63);
    int s, d; edge_ends(e, src, dst, s, d);
    float coeff = alpha[e] / ssum[d];
    atomicAdd(&out[(size_t)d * EMB + c], xl[(size_t)s * EMB + c] * coeff);
}

// ---------------- BatchNorm (train-mode, biased var) + ReLU ----------------
// blockDim.x must be a multiple of C (C in {64,128}); per-block LDS reduce.
__global__ void k_bn_stats(const float* __restrict__ h, int n, int C,
                           float* __restrict__ sum, float* __restrict__ sq) {
    __shared__ float ls[256], lq[256];
    int tid = threadIdx.x;
    int rowsPer = blockDim.x / C;
    int c = tid % C;
    float s = 0.f, q = 0.f;
    for (int r = blockIdx.x * rowsPer + tid / C; r < n; r += gridDim.x * rowsPer) {
        float v = h[(size_t)r * C + c];
        s += v; q += v * v;
    }
    ls[tid] = s; lq[tid] = q;
    __syncthreads();
    if (tid < C) {
        for (int t = tid + C; t < blockDim.x; t += C) { s += ls[t]; q += lq[t]; }
        atomicAdd(&sum[tid], s);
        atomicAdd(&sq[tid], q);
    }
}

// NOTE: the GAT output bias (bias1/bias2) is a per-channel constant added BEFORE
// BatchNorm; BN output is exactly invariant to per-channel constant shifts
// (mean absorbs it, var unchanged), so it is omitted — mathematically identical.
__global__ void k_bn_apply_relu(const float* __restrict__ h, const float* __restrict__ sum,
                                const float* __restrict__ sq, const float* __restrict__ gamma,
                                const float* __restrict__ beta, float* __restrict__ out,
                                int n, int C) {
    size_t idx = (size_t)blockIdx.x * blockDim.x + threadIdx.x;
    if (idx >= (size_t)n * C) return;
    int c = (int)(idx % C);
    float inv_n = 1.0f / (float)n;
    float mu = sum[c] * inv_n;
    float var = sq[c] * inv_n - mu * mu;
    float v = (h[idx] - mu) * rsqrtf(var + BN_EPS) * gamma[c] + beta[c];
    out[idx] = v > 0.f ? v : 0.f;
}

// ---------------- pooling + classifier head ----------------
__global__ void k_pool(const float* __restrict__ emb, const int* __restrict__ batch,
                       float* __restrict__ gsum, float* __restrict__ gcnt) {
    size_t idx = (size_t)blockIdx.x * blockDim.x + threadIdx.x;
    if (idx >= (size_t)N_NODES * EMB) return;
    int i = (int)(idx / EMB), c = (int)(idx % EMB);
    int g = batch[i];
    atomicAdd(&gsum[(size_t)g * EMB + c], emb[idx]);
    if (c == 0) atomicAdd(&gcnt[g], 1.0f);
}

__global__ void k_head(const float* __restrict__ gsum, const float* __restrict__ gcnt,
                       const float* __restrict__ Wc, const float* __restrict__ bc,
                       float* __restrict__ out_logits, float* __restrict__ out_demb) {
    __shared__ float demb[EMB];
    int g = blockIdx.x, c = threadIdx.x;
    float cnt = fmaxf(gcnt[g], 1.f);
    float v = gsum[(size_t)g * EMB + c] / cnt;
    demb[c] = v;
    out_demb[(size_t)g * EMB + c] = v;
    __syncthreads();
    if (c < NCLS) {
        float acc = bc[c];
        #pragma unroll 8
        for (int k = 0; k < EMB; ++k) acc += demb[k] * Wc[k * NCLS + c];
        out_logits[g * NCLS + c] = acc;
    }
}

// ---------------- launcher ----------------
static inline unsigned cdiv(size_t a, size_t b) { return (unsigned)((a + b - 1) / b); }

extern "C" void kernel_launch(void* const* d_in, const int* in_sizes, int n_in,
                              void* d_out, int out_size, void* d_ws, size_t ws_size,
                              hipStream_t stream) {
    (void)in_sizes; (void)n_in; (void)out_size; (void)ws_size;

    // inputs (setup_inputs dict order)
    const float* x      = (const float*)d_in[0];
    const int*   ei     = (const int*)d_in[1];
    const float* eattr  = (const float*)d_in[2];
    const int*   batch  = (const int*)d_in[3];
    const float* Wl1    = (const float*)d_in[4];
    const float* bl1    = (const float*)d_in[5];
    const float* Wr1    = (const float*)d_in[6];
    const float* br1    = (const float*)d_in[7];
    const float* We1    = (const float*)d_in[8];
    const float* att1   = (const float*)d_in[9];
    /* bias1 = d_in[10]  (no-op under following BN, see note) */
    const float* Wl2    = (const float*)d_in[11];
    const float* bl2    = (const float*)d_in[12];
    const float* Wr2    = (const float*)d_in[13];
    const float* br2    = (const float*)d_in[14];
    const float* We2    = (const float*)d_in[15];
    const float* att2   = (const float*)d_in[16];
    /* bias2 = d_in[17]  (no-op under following BN) */
    const float* gamma1 = (const float*)d_in[18];
    const float* beta1  = (const float*)d_in[19];
    const float* gamma2 = (const float*)d_in[20];
    const float* beta2  = (const float*)d_in[21];
    const float* Wc     = (const float*)d_in[22];
    const float* bc     = (const float*)d_in[23];

    const int* src = ei;             // edge_index[0]
    const int* dst = ei + N_EDGES;   // edge_index[1]

    // output layout: logits[50*6] | domain_emb[50*64] | node_emb[50000*64]
    float* out_logits = (float*)d_out;
    float* out_demb   = out_logits + G_GR * NCLS;
    float* node_emb   = out_demb + G_GR * EMB;

    // workspace carve-up (floats)
    float* ws    = (float*)d_ws;
    float* deg   = ws;                                   // N
    float* easum = deg + N_NODES;                        // N
    float* ea2   = easum + N_NODES;                      // E2
    float* xl1   = ea2 + E2;                             // N*HID   (reused: h2)
    float* xr1   = xl1 + (size_t)N_NODES * HID;          // N*HID   (reused: hb1)
    float* alpha1= xr1 + (size_t)N_NODES * HID;          // E2*H1   (reused: xl2,xr2)
    float* m1    = alpha1 + (size_t)E2 * H1;             // N*H1
    float* s1    = m1 + (size_t)N_NODES * H1;            // N*H1
    float* h1    = s1 + (size_t)N_NODES * H1;            // N*HID
    float* bn1s  = h1 + (size_t)N_NODES * HID;           // HID
    float* bn1q  = bn1s + HID;                           // HID
    float* alpha2= bn1q + HID;                           // E2
    float* m2    = alpha2 + E2;                          // N
    float* s2    = m2 + N_NODES;                         // N
    float* gsum  = s2 + N_NODES;                         // G*EMB
    float* gcnt  = gsum + (size_t)G_GR * EMB;            // G
    float* bn2s  = gcnt + G_GR;                          // EMB
    float* bn2q  = bn2s + EMB;                           // EMB
    // aliases (safe: donor's last read precedes alias's first write)
    float* hb1 = xr1;                                    // post-BN layer-1 features
    float* xl2 = alpha1;
    float* xr2 = alpha1 + (size_t)N_NODES * EMB;
    float* h2  = xl1;

    const int B = 256;

    // ---- self loops ----
    k_fill<<<cdiv(2 * N_NODES, B), B, 0, stream>>>(deg, 0.f, (size_t)2 * N_NODES); // deg+easum
    k_deg_sum<<<cdiv(N_EDGES, B), B, 0, stream>>>(dst, eattr, deg, easum, ea2);
    k_loop_attr<<<cdiv(N_NODES, B), B, 0, stream>>>(deg, easum, ea2);

    // ---- layer 1 projections (WMMA, 64 rows/block) ----
    k_gemm_wmma<F_IN, HID><<<cdiv(N_NODES, 64), HID * 2, 0, stream>>>(x, Wl1, bl1, xl1, N_NODES);
    k_gemm_wmma<F_IN, HID><<<cdiv(N_NODES, 64), HID * 2, 0, stream>>>(x, Wr1, br1, xr1, N_NODES);

    // ---- layer 1 attention + aggregate ----
    k_fill<<<cdiv((size_t)N_NODES * H1, B), B, 0, stream>>>(m1, -1e30f, (size_t)N_NODES * H1);
    k_fill<<<cdiv((size_t)N_NODES * H1, B), B, 0, stream>>>(s1, 0.f, (size_t)N_NODES * H1);
    k_fill<<<cdiv((size_t)N_NODES * HID, B), B, 0, stream>>>(h1, 0.f, (size_t)N_NODES * HID);
    k_logits1<<<cdiv((size_t)E2 * H1, B), B, 0, stream>>>(xl1, xr1, ea2, src, dst, We1, att1, alpha1, m1);
    k_exp1<<<cdiv((size_t)E2 * H1, B), B, 0, stream>>>(src, dst, m1, alpha1, s1);
    k_scatter1<<<cdiv((size_t)E2 * HID, B), B, 0, stream>>>(src, dst, alpha1, s1, xl1, h1);

    // ---- BN1 + ReLU -> hb1 (reuses xr1) ----
    k_fill<<<1, B, 0, stream>>>(bn1s, 0.f, (size_t)2 * HID);
    k_bn_stats<<<1024, B, 0, stream>>>(h1, N_NODES, HID, bn1s, bn1q);
    k_bn_apply_relu<<<cdiv((size_t)N_NODES * HID, B), B, 0, stream>>>(h1, bn1s, bn1q, gamma1, beta1,
                                                                      hb1, N_NODES, HID);

    // ---- layer 2 projections (WMMA; xl2/xr2 reuse alpha1) ----
    k_gemm_wmma<HID, EMB><<<cdiv(N_NODES, 64), EMB * 2, 0, stream>>>(hb1, Wl2, bl2, xl2, N_NODES);
    k_gemm_wmma<HID, EMB><<<cdiv(N_NODES, 64), EMB * 2, 0, stream>>>(hb1, Wr2, br2, xr2, N_NODES);

    // ---- layer 2 attention + aggregate (h2 reuses xl1) ----
    k_fill<<<cdiv(N_NODES, B), B, 0, stream>>>(m2, -1e30f, (size_t)N_NODES);
    k_fill<<<cdiv(N_NODES, B), B, 0, stream>>>(s2, 0.f, (size_t)N_NODES);
    k_fill<<<cdiv((size_t)N_NODES * EMB, B), B, 0, stream>>>(h2, 0.f, (size_t)N_NODES * EMB);
    k_logits2<<<cdiv(E2, B), B, 0, stream>>>(xl2, xr2, ea2, src, dst, We2, att2, alpha2, m2);
    k_exp2<<<cdiv(E2, B), B, 0, stream>>>(src, dst, m2, alpha2, s2);
    k_scatter2<<<cdiv((size_t)E2 * EMB, B), B, 0, stream>>>(src, dst, alpha2, s2, xl2, h2);

    // ---- BN2 + ReLU -> node_emb (into d_out) ----
    k_fill<<<1, B, 0, stream>>>(bn2s, 0.f, (size_t)2 * EMB);
    k_bn_stats<<<1024, B, 0, stream>>>(h2, N_NODES, EMB, bn2s, bn2q);
    k_bn_apply_relu<<<cdiv((size_t)N_NODES * EMB, B), B, 0, stream>>>(h2, bn2s, bn2q, gamma2, beta2,
                                                                      node_emb, N_NODES, EMB);

    // ---- mean pool + classifier ----
    k_fill<<<cdiv((size_t)G_GR * EMB + G_GR, B), B, 0, stream>>>(gsum, 0.f, (size_t)G_GR * EMB + G_GR);
    k_pool<<<cdiv((size_t)N_NODES * EMB, B), B, 0, stream>>>(node_emb, batch, gsum, gcnt);
    k_head<<<G_GR, EMB, 0, stream>>>(gsum, gcnt, Wc, bc, out_logits, out_demb);
}
